// InteractiveEncoder_31421980738244
// MI455X (gfx1250) — compile-verified, matrix operand order checked by reading, not run on hardware
//
#include <hip/hip_runtime.h>

// ---------------- types ----------------
typedef __attribute__((ext_vector_type(16))) _Float16 v16h;
typedef __attribute__((ext_vector_type(8)))  _Float16 h8;   // 16B, natural align 16
typedef __attribute__((ext_vector_type(8)))  float    v8f;

// ---------------- problem constants ----------------
static constexpr int kIN  = 32;
static constexpr int kOUT = 16;
static constexpr int kL   = 10;

// ---------------- LDS layout (units: _Float16 halfs) ----------------
// Shared weight tiles stored [N][K] f16, K-stride padded for conflict-free
// B-fragment loads (row stride in banks == 4 mod 32).
static constexpr int OFF_WG  = 0;                    // [256][136]  (W_ih|W_hh merged, K=128)
static constexpr int S_WG    = 256 * 136;
static constexpr int OFF_WUP = OFF_WG + S_WG;        // [512][72]   (K=64)
static constexpr int S_WUP   = 512 * 72;
static constexpr int OFF_WD1 = OFF_WUP + S_WUP;      // [128][136]
static constexpr int S_WD1   = 128 * 136;
static constexpr int OFF_WD2 = OFF_WD1 + S_WD1;      // [32][136]
static constexpr int S_WD2   = 32 * 136;
static constexpr int OFF_WP  = OFF_WD2 + S_WD2;      // [64][72]
static constexpr int S_WP    = 64 * 72;
static constexpr int OFF_WO1 = OFF_WP + S_WP;        // [128][72]
static constexpr int S_WO1   = 128 * 72;
// float bias region
static constexpr int OFF_BIAS = OFF_WO1 + S_WO1;     // float[1248]
static constexpr int BF_BG  = 0;     // 256  (b_ih + b_hh)
static constexpr int BF_BUP = 256;   // 512
static constexpr int BF_BD1 = 768;   // 128
static constexpr int BF_BD2 = 896;   // 32
static constexpr int BF_BP  = 928;   // 64
static constexpr int BF_BO1 = 992;   // 128
static constexpr int BF_WO2 = 1120;  // 128
static constexpr int NBIAS  = 1248;
// per-team (16 samples, 4 waves) staging
static constexpr int OFF_TEAM = OFF_BIAS + 2 * NBIAS;
static constexpr int T_ACT = 0;          // [16][136]  x|h  (x: cols 0..63, h: 64..127)
static constexpr int T_U   = 2176;       // [64][136]  u / d
static constexpr int T_PI  = 10880;      // [16][136]  pi  ([s][p*32+i])
static constexpr int T_PO  = 13056;      // [16][72]   po  ([s][p*16+o])
static constexpr int T_LW  = 14208;      // [16][16 o (stride 40)][32]  lin_w f16 frags
static constexpr int TEAM_HALFS = 24448;
static constexpr int NTEAMS = 2;         // 8 waves / 256 threads per block
static constexpr int SMEM_HALFS = OFF_TEAM + NTEAMS * TEAM_HALFS; // 158656 -> 317312 B

// ---------------- device helpers ----------------
__device__ __forceinline__ v16h ldfrag(const _Float16* p) {
    h8 lo = *(const h8*)p;
    h8 hi = *(const h8*)(p + 8);
    return __builtin_shufflevector(lo, hi, 0,1,2,3,4,5,6,7,8,9,10,11,12,13,14,15);
}

__device__ __forceinline__ v8f wmma_f16(v16h a, v16h b, v8f c) {
    return __builtin_amdgcn_wmma_f32_16x16x32_f16(false, a, false, b, (short)0, c,
                                                  false, false);
}

__device__ __forceinline__ v8f splat8(float x) {
    v8f v;
#pragma unroll
    for (int i = 0; i < 8; ++i) v[i] = x;
    return v;
}

__device__ __forceinline__ float fast_sigmoid(float x) {
    return __builtin_amdgcn_rcpf(1.0f + __expf(-x));
}
__device__ __forceinline__ float fast_tanh(float x) {   // 2*sigmoid(2x) - 1
    return __builtin_fmaf(2.0f, fast_sigmoid(2.0f * x), -1.0f);
}

// ---------------- kernel ----------------
__global__ void __launch_bounds__(256)
ie_rollout_kernel(const float* __restrict__ g_lin_w, const float* __restrict__ g_lin_b,
                  const float* __restrict__ g_Wih,   const float* __restrict__ g_Whh,
                  const float* __restrict__ g_bih,   const float* __restrict__ g_bhh,
                  const float* __restrict__ g_bos,   const float* __restrict__ g_eos,
                  const float* __restrict__ g_Wup,   const float* __restrict__ g_bup,
                  const float* __restrict__ g_Wd1,   const float* __restrict__ g_bd1,
                  const float* __restrict__ g_Wd2,   const float* __restrict__ g_bd2,
                  const float* __restrict__ g_Wp,    const float* __restrict__ g_bp,
                  const float* __restrict__ g_Wo1,   const float* __restrict__ g_bo1,
                  const float* __restrict__ g_Wo2,   const float* __restrict__ g_bo2,
                  float* __restrict__ g_out)
{
    extern __shared__ _Float16 smem[];
    float* bias = (float*)(smem + OFF_BIAS);

    const int tid   = threadIdx.x;
    const int lane  = tid & 31;
    const int wv    = tid >> 5;
    const int team  = wv >> 2;     // 2 teams of 4 waves; each team owns 16 samples
    const int wq    = wv & 3;      // wave role within team
    const int laneM = lane & 15;   // M row / N col within fragment
    const int laneH = lane >> 4;   // K-half selector

    // ---- cooperative shared-weight staging: f32 global -> f16 LDS [N][K] ----
    for (int i = tid; i < 256 * 128; i += 256) {                   // gates: [W_ih | W_hh]
        int n = i >> 7, k = i & 127;
        float v = (k < 64) ? g_Wih[n * 64 + k] : g_Whh[n * 64 + (k - 64)];
        smem[OFF_WG + n * 136 + k] = (_Float16)v;
    }
    for (int i = tid; i < 512 * 64; i += 256) { int n = i >> 6, k = i & 63;
        smem[OFF_WUP + n * 72 + k] = (_Float16)g_Wup[i]; }
    for (int i = tid; i < 128 * 128; i += 256) { int n = i >> 7, k = i & 127;
        smem[OFF_WD1 + n * 136 + k] = (_Float16)g_Wd1[i]; }
    for (int i = tid; i < 32 * 128; i += 256) { int n = i >> 7, k = i & 127;
        smem[OFF_WD2 + n * 136 + k] = (_Float16)g_Wd2[i]; }
    for (int i = tid; i < 64 * 64; i += 256) { int n = i >> 6, k = i & 63;
        smem[OFF_WP + n * 72 + k] = (_Float16)g_Wp[i]; }
    for (int i = tid; i < 128 * 64; i += 256) { int n = i >> 6, k = i & 63;
        smem[OFF_WO1 + n * 72 + k] = (_Float16)g_Wo1[i]; }
    if (tid < 256) bias[BF_BG + tid] = g_bih[tid] + g_bhh[tid];
    for (int i = tid; i < 512; i += 256) bias[BF_BUP + i] = g_bup[i];
    if (tid < 128) bias[BF_BD1 + tid] = g_bd1[tid];
    if (tid < 32)  bias[BF_BD2 + tid] = g_bd2[tid];
    if (tid < 64)  bias[BF_BP + tid]  = g_bp[tid];
    if (tid < 128) bias[BF_BO1 + tid] = g_bo1[tid];
    if (tid < 128) bias[BF_WO2 + tid] = g_Wo2[tid];

    // ---- per-team staging pointers ----
    _Float16* ts   = smem + OFF_TEAM + team * TEAM_HALFS;
    _Float16* actA = ts + T_ACT;
    _Float16* uB   = ts + T_U;
    _Float16* piB  = ts + T_PI;
    _Float16* poB  = ts + T_PO;
    _Float16* lwB  = ts + T_LW;
    const int S0   = (blockIdx.x * NTEAMS + team) * 16;
    const int t128 = wq * 32 + lane;   // 0..127 within team

    // per-sample probe weights -> f16 B-fragment layout [s][o (stride 40)][i]
    for (int i = t128; i < 16 * 16 * 32; i += 128) {
        int s = i >> 9, o = (i >> 5) & 15, ii = i & 31;
        lwB[s * 640 + o * 40 + ii] =
            (_Float16)g_lin_w[(size_t)(S0 + s) * (kOUT * kIN) + o * kIN + ii];
    }
    // per-sample probe bias in registers: wave wq handles samples wq*4..wq*4+3
    float lb[4];
#pragma unroll
    for (int s4 = 0; s4 < 4; ++s4) lb[s4] = g_lin_b[(S0 + wq * 4 + s4) * kOUT + laneM];

    // init x = bos (broadcast), h = 0
    for (int i = t128; i < 16 * 64; i += 128) {
        int r = i >> 6, c = i & 63;
        actA[r * 136 + c]      = (_Float16)g_bos[c];
        actA[r * 136 + 64 + c] = (_Float16)0.0f;
    }

    // LSTM cell state for this wave's hidden tile (cols 16*wq..16*wq+15), C layout
    v8f cst = splat8(0.0f);

    __syncthreads();   // weights + staging visible

    // ---- one LSTM cell step: reads actA [x|h]; this wave computes hidden tile wq ----
    auto lstm_step = [&]() {
        v16h a[4];
#pragma unroll
        for (int k = 0; k < 4; ++k)
            a[k] = ldfrag(actA + laneM * 136 + k * 32 + laneH * 16);
        v8f gi = splat8(bias[BF_BG + (wq)      * 16 + laneM]);
        v8f gf = splat8(bias[BF_BG + (4 + wq)  * 16 + laneM]);
        v8f gg = splat8(bias[BF_BG + (8 + wq)  * 16 + laneM]);
        v8f go = splat8(bias[BF_BG + (12 + wq) * 16 + laneM]);
#pragma unroll
        for (int k = 0; k < 4; ++k) {
            gi = wmma_f16(a[k], ldfrag(smem + OFF_WG + ((wq)      * 16 + laneM) * 136 + k * 32 + laneH * 16), gi);
            gf = wmma_f16(a[k], ldfrag(smem + OFF_WG + ((4 + wq)  * 16 + laneM) * 136 + k * 32 + laneH * 16), gf);
            gg = wmma_f16(a[k], ldfrag(smem + OFF_WG + ((8 + wq)  * 16 + laneM) * 136 + k * 32 + laneH * 16), gg);
            go = wmma_f16(a[k], ldfrag(smem + OFF_WG + ((12 + wq) * 16 + laneM) * 136 + k * 32 + laneH * 16), go);
        }
#pragma unroll
        for (int r = 0; r < 8; ++r) {
            float iv = fast_sigmoid(gi[r]);
            float fv = fast_sigmoid(gf[r]);
            float tv = fast_tanh(gg[r]);
            float ov = fast_sigmoid(go[r]);
            float c2 = fv * cst[r] + iv * tv;
            cst[r] = c2;
            float hv = ov * fast_tanh(c2);
            actA[(r + 8 * laneH) * 136 + 64 + wq * 16 + laneM] = (_Float16)hv;
        }
    };

    // ================= probing rollout =================
    for (int step = 0; step < kL; ++step) {
        lstm_step();
        __syncthreads();                       // h complete (all 4 tiles)

        // ---- proj_up: u = relu(h @ Wup^T + bup) -> uB (this wave: 8 N-tiles) ----
        {
            v16h h0 = ldfrag(actA + laneM * 136 + 64 + laneH * 16);
            v16h h1 = ldfrag(actA + laneM * 136 + 96 + laneH * 16);
#pragma unroll 2
            for (int nn = 0; nn < 8; ++nn) {   // unroll 2: independent WMMA chains
                int n = wq * 8 + nn;
                v8f acc = splat8(bias[BF_BUP + n * 16 + laneM]);
                acc = wmma_f16(h0, ldfrag(smem + OFF_WUP + (n * 16 + laneM) * 72 + 0  + laneH * 16), acc);
                acc = wmma_f16(h1, ldfrag(smem + OFF_WUP + (n * 16 + laneM) * 72 + 32 + laneH * 16), acc);
                int p = n >> 3, ec = (n & 7) * 16 + laneM;
#pragma unroll
                for (int r = 0; r < 8; ++r) {
                    float v = acc[r]; v = v > 0.0f ? v : 0.0f;
                    uB[((r + 8 * laneH) * 4 + p) * 136 + ec] = (_Float16)v;
                }
            }
        }
        __syncthreads();                       // uB complete

        // ---- decoder + pi on this wave's M-tile (m = wq) ----
        {
            _Float16* ub = uB + wq * 16 * 136;
            v16h ua[4];
#pragma unroll
            for (int k = 0; k < 4; ++k)
                ua[k] = ldfrag(ub + laneM * 136 + k * 32 + laneH * 16);
#pragma unroll 2
            for (int n = 0; n < 8; ++n) {      // unroll 2: independent WMMA chains
                v8f acc = splat8(bias[BF_BD1 + n * 16 + laneM]);
#pragma unroll
                for (int k = 0; k < 4; ++k)
                    acc = wmma_f16(ua[k], ldfrag(smem + OFF_WD1 + (n * 16 + laneM) * 136 + k * 32 + laneH * 16), acc);
#pragma unroll
                for (int r = 0; r < 8; ++r) {
                    float v = acc[r]; v = v > 0.0f ? v : 0.0f;
                    ub[(r + 8 * laneH) * 136 + n * 16 + laneM] = (_Float16)v;
                }
            }
            v16h da[4];
#pragma unroll
            for (int k = 0; k < 4; ++k)
                da[k] = ldfrag(ub + laneM * 136 + k * 32 + laneH * 16);
#pragma unroll
            for (int n = 0; n < 2; ++n) {
                v8f acc = splat8(bias[BF_BD2 + n * 16 + laneM]);
#pragma unroll
                for (int k = 0; k < 4; ++k)
                    acc = wmma_f16(da[k], ldfrag(smem + OFF_WD2 + (n * 16 + laneM) * 136 + k * 32 + laneH * 16), acc);
#pragma unroll
                for (int r = 0; r < 8; ++r) {
                    float v = fast_sigmoid(acc[r]);
                    int g = wq * 16 + r + 8 * laneH;     // rows = samples wq*4..wq*4+3
                    int s = g >> 2, p = g & 3;
                    piB[s * 136 + p * 32 + n * 16 + laneM] = (_Float16)v;
                }
            }
        }

        // ---- vfunc: per-sample probe (samples wq*4..wq*4+3, written above by this wave) ----
#pragma unroll 2
        for (int s4 = 0; s4 < 4; ++s4) {       // unroll 2: independent WMMAs
            int s = wq * 4 + s4;
            v16h ap = ldfrag(piB + s * 136 + laneM * 32 + laneH * 16);  // A = pi[s] (rows>=4 ignored)
            v16h bw = ldfrag(lwB + s * 640 + laneM * 40 + laneH * 16);  // B = lin_w[s]^T
            v8f acc = splat8(lb[s4]);
            acc = wmma_f16(ap, bw, acc);
            if (laneH == 0) {
#pragma unroll
                for (int r = 0; r < 4; ++r)
                    poB[s * 72 + r * 16 + laneM] = (_Float16)fast_sigmoid(acc[r]);
            }
        }
        __syncthreads();                       // poB complete

        // ---- probing_output_transform: x_next = sigmoid(po @ Wp^T + bp), N-tile wq ----
        {
            v16h p0 = ldfrag(poB + laneM * 72 + laneH * 16);
            v16h p1 = ldfrag(poB + laneM * 72 + 32 + laneH * 16);
            v8f acc = splat8(bias[BF_BP + wq * 16 + laneM]);
            acc = wmma_f16(p0, ldfrag(smem + OFF_WP + (wq * 16 + laneM) * 72 + 0  + laneH * 16), acc);
            acc = wmma_f16(p1, ldfrag(smem + OFF_WP + (wq * 16 + laneM) * 72 + 32 + laneH * 16), acc);
#pragma unroll
            for (int r = 0; r < 8; ++r)
                actA[(r + 8 * laneH) * 136 + wq * 16 + laneM] = (_Float16)fast_sigmoid(acc[r]);
        }
        __syncthreads();                       // actA [x|h] complete for next step
    }

    // ================= tail: two more LSTM steps + output head =================
    lstm_step();                               // x = last xn
    for (int i = t128; i < 16 * 64; i += 128) {  // x = eos (disjoint cols from h)
        int r = i >> 6, c = i & 63;
        actA[r * 136 + c] = (_Float16)g_eos[c];
    }
    __syncthreads();
    lstm_step();
    __syncthreads();

    // r = relu(h @ Wo1^T + bo1) @ Wo2^T + bo2   (one wave per team)
    if (wq == 0) {
        v16h h0 = ldfrag(actA + laneM * 136 + 64 + laneH * 16);
        v16h h1 = ldfrag(actA + laneM * 136 + 96 + laneH * 16);
        float racc[8];
#pragma unroll
        for (int r = 0; r < 8; ++r) racc[r] = 0.0f;
#pragma unroll 2
        for (int n = 0; n < 8; ++n) {          // unroll 2: independent WMMA chains
            v8f acc = splat8(bias[BF_BO1 + n * 16 + laneM]);
            acc = wmma_f16(h0, ldfrag(smem + OFF_WO1 + (n * 16 + laneM) * 72 + 0  + laneH * 16), acc);
            acc = wmma_f16(h1, ldfrag(smem + OFF_WO1 + (n * 16 + laneM) * 72 + 32 + laneH * 16), acc);
            float w2 = bias[BF_WO2 + n * 16 + laneM];
#pragma unroll
            for (int r = 0; r < 8; ++r) {
                float v = acc[r]; v = v > 0.0f ? v : 0.0f;
                racc[r] += v * w2;
            }
        }
        float b2 = g_bo2[0];
#pragma unroll
        for (int r = 0; r < 8; ++r) {          // reduce 16 lanes per half-wave
            float sum = racc[r];
            sum += __shfl_xor(sum, 1, 32);
            sum += __shfl_xor(sum, 2, 32);
            sum += __shfl_xor(sum, 4, 32);
            sum += __shfl_xor(sum, 8, 32);
            if (laneM == 0) g_out[S0 + r + 8 * laneH] = sum + b2;
        }
    }
}

// ---------------- host launcher ----------------
extern "C" void kernel_launch(void* const* d_in, const int* in_sizes, int n_in,
                              void* d_out, int out_size, void* d_ws, size_t ws_size,
                              hipStream_t stream) {
    (void)n_in; (void)d_ws; (void)ws_size; (void)out_size;
    const float* lin_w = (const float*)d_in[0];
    const float* lin_b = (const float*)d_in[1];
    const float* Wih   = (const float*)d_in[2];
    const float* Whh   = (const float*)d_in[3];
    const float* bih   = (const float*)d_in[4];
    const float* bhh   = (const float*)d_in[5];
    const float* bos   = (const float*)d_in[6];
    const float* eos   = (const float*)d_in[7];
    // d_in[8] = probing_encodings: unused (STATIC_W == 0)
    const float* Wup = (const float*)d_in[9];
    const float* bup = (const float*)d_in[10];
    const float* Wd1 = (const float*)d_in[11];
    const float* bd1 = (const float*)d_in[12];
    const float* Wd2 = (const float*)d_in[13];
    const float* bd2 = (const float*)d_in[14];
    const float* Wp  = (const float*)d_in[15];
    const float* bp  = (const float*)d_in[16];
    const float* Wo1 = (const float*)d_in[17];
    const float* bo1 = (const float*)d_in[18];
    const float* Wo2 = (const float*)d_in[19];
    const float* bo2 = (const float*)d_in[20];

    const int BS     = in_sizes[0] / (kOUT * kIN);   // 16384
    const int blocks = BS / (16 * NTEAMS);           // 16 samples/team, 2 teams/block
    const size_t shmem = (size_t)SMEM_HALFS * sizeof(_Float16); // 317312 B (<= 320KB WGP LDS)

    // Opt in to >64KB dynamic LDS (host-side attr set; deterministic, capture-safe).
    (void)hipFuncSetAttribute(reinterpret_cast<const void*>(ie_rollout_kernel),
                              hipFuncAttributeMaxDynamicSharedMemorySize, (int)shmem);

    ie_rollout_kernel<<<blocks, 256, shmem, stream>>>(
        lin_w, lin_b, Wih, Whh, bih, bhh, bos, eos,
        Wup, bup, Wd1, bd1, Wd2, bd2, Wp, bp, Wo1, bo1, Wo2, bo2,
        (float*)d_out);
}
